// NeuRayIndependentToken_30580167147793
// MI455X (gfx1250) — matrix operand order repaired and checked by reading, not compiled, not run on hardware
//
#include <hip/hip_runtime.h>
#include <hip/hip_bf16.h>
#include <stdint.h>

typedef __attribute__((ext_vector_type(16))) __bf16 v16bf;
typedef __attribute__((ext_vector_type(8)))  float  v8f;

union FragBF { v16bf v; unsigned int u[8]; };

__device__ __forceinline__ unsigned short f2bf(float f) {
    unsigned int u = __float_as_uint(f);
    u += 0x7fffu + ((u >> 16) & 1u);          // round-to-nearest-even
    return (unsigned short)(u >> 16);
}
__device__ __forceinline__ unsigned int pack2(float a, float b) {
    return (unsigned int)f2bf(a) | ((unsigned int)f2bf(b) << 16);
}
__device__ __forceinline__ float eluf(float x) { return x > 0.f ? x : (__expf(x) - 1.f); }
__device__ __forceinline__ float sigm(float x) { return 1.f / (1.f + __expf(-x)); }

// ---------------------------------------------------------------------------
// Kernel W: repack if0_w (2048x512) and if1_w (512x32) f32 -> bf16 WMMA
// B-fragment layout.  Per fragment (one 32x16 K x N tile): lane holds column
// n = lane&15, K-range klo = (lane>>4)*16, 16 consecutive bf16 (pairs per
// dword) -> 8 dwords contiguous per lane.
// ---------------------------------------------------------------------------
__global__ __launch_bounds__(256) void prep_weights(
    const float* __restrict__ w0, const float* __restrict__ w1,
    unsigned int* __restrict__ wsB0, unsigned int* __restrict__ wsB1)
{
    int tid  = blockIdx.x * blockDim.x + threadIdx.x;
    int lane = tid & 31;
    int n    = lane & 15;
    int sel  = (lane >> 4) & 1;
    if (tid < 65536) {                       // if0: 32 n-tiles x 64 k-chunks
        int frag = tid >> 5;                 // 0..2047  (= nt*64 + kcg)
        int nt   = frag >> 6;
        int kcg  = frag & 63;
        int ng   = nt * 16 + n;
        int k0   = kcg * 32 + sel * 16;
        unsigned int* dst = wsB0 + (size_t)(frag * 32 + lane) * 8;
        #pragma unroll
        for (int j = 0; j < 8; ++j) {
            float a = w0[(size_t)(k0 + 2 * j) * 512 + ng];
            float b = w0[(size_t)(k0 + 2 * j + 1) * 512 + ng];
            dst[j] = pack2(a, b);
        }
    } else if (tid < 66560) {                // if1: 2 n-tiles x 16 k-chunks
        int t2   = tid - 65536;
        int frag = t2 >> 5;                  // 0..31 (= nt2*16 + kc)
        int nt2  = frag >> 4;
        int kc   = frag & 15;
        int ng   = nt2 * 16 + n;
        int k0   = kc * 32 + sel * 16;
        unsigned int* dst = wsB1 + (size_t)(frag * 32 + lane) * 8;
        #pragma unroll
        for (int j = 0; j < 8; ++j) {
            float a = w1[(k0 + 2 * j) * 32 + ng];
            float b = w1[(k0 + 2 * j + 1) * 32 + ng];
            dst[j] = pack2(a, b);
        }
    }
}

// ---------------------------------------------------------------------------
// Kernel G: per block, 64 token-view rows (4 M-subtiles of 16).
//   h   = elu(bott(64x2048) @ if0_w + if0_b)        (bf16 WMMA, N=512)
//   out = h @ if1_w + if1_b + direction_feat        (bf16 WMMA, N=32) -> P
// Each of 8 waves owns 4 N-tiles x 4 M-subtiles (128 acc VGPRs); every B
// fragment load feeds 4 WMMAs.  K staged in 8 LDS chunks of 256 (padded
// stride, == 4 mod 64 dwords).  A LDS buffer is reused as the bf16 hidden
// staging area for the second layer, processed in 4 row-groups of 16.
// ---------------------------------------------------------------------------
#define AW 132   // dword stride of A LDS row (264 bf16, holds 256)
#define HW 260   // dword stride of h LDS row (520 bf16, holds 512)

__global__ __launch_bounds__(256) void big_gemm(
    const float* __restrict__ bott, const float* __restrict__ ray_diff,
    const float* __restrict__ rd0_w, const float* __restrict__ rd0_b,
    const float* __restrict__ rd1_w, const float* __restrict__ rd1_b,
    const float* __restrict__ if0_b, const float* __restrict__ if1_b,
    const unsigned int* __restrict__ wsB0, const unsigned int* __restrict__ wsB1,
    float* __restrict__ P)
{
    __shared__ unsigned int AuU[64 * AW];   // 33.8 KB  bf16 A chunk / h staging
    __shared__ float dirf[64 * 32];         //  8.0 KB  direction features

    const int tid  = threadIdx.x;
    const int wave = tid >> 5;
    const int lane = tid & 31;
    const int mrow = lane & 15;
    const int sel  = (lane >> 4) & 1;

    // direction feature: 4 -> 16 (elu) -> 32 (elu), one thread per row
    if (tid < 64) {
        size_t g = (size_t)blockIdx.x * 64 + tid;
        const float* rp = ray_diff + g * 4;
        float x0 = rp[0], x1 = rp[1], x2 = rp[2], x3 = rp[3];
        float h16[16];
        #pragma unroll
        for (int j = 0; j < 16; ++j)
            h16[j] = eluf(rd0_b[j] + x0 * rd0_w[j] + x1 * rd0_w[16 + j] +
                          x2 * rd0_w[32 + j] + x3 * rd0_w[48 + j]);
        for (int n = 0; n < 32; ++n) {
            float a = rd1_b[n];
            #pragma unroll
            for (int j = 0; j < 16; ++j) a += h16[j] * rd1_w[j * 32 + n];
            dirf[tid * 32 + n] = eluf(a);
        }
    }

    v8f acc[4][4];                           // [m-subtile][n-tile]
    #pragma unroll
    for (int ms = 0; ms < 4; ++ms)
        #pragma unroll
        for (int q = 0; q < 4; ++q) acc[ms][q] = (v8f){0, 0, 0, 0, 0, 0, 0, 0};

    for (int ch = 0; ch < 8; ++ch) {         // K chunks of 256
        // cooperative load: 64 rows x 256 cols f32 -> bf16 LDS
        #pragma unroll
        for (int i = 0; i < 16; ++i) {
            int e   = i * 256 + tid;
            int row = e >> 6;                // /64 float4 per row
            int c4  = e & 63;
            size_t g = (size_t)blockIdx.x * 64 + row;
            const float4 f =
                *(const float4*)(bott + g * 2048 + (size_t)ch * 256 + (size_t)c4 * 4);
            unsigned int* dst = AuU + row * AW + c4 * 2;
            dst[0] = pack2(f.x, f.y);
            dst[1] = pack2(f.z, f.w);
        }
        __syncthreads();

        for (int kc = 0; kc < 8; ++kc) {     // 32-wide K steps inside chunk
            FragBF a[4];                     // A frags, one per M-subtile
            const int kbase = kc * 32 + sel * 16;
            #pragma unroll
            for (int ms = 0; ms < 4; ++ms) {
                const int rbase = (ms * 16 + mrow) * AW;
                #pragma unroll
                for (int j = 0; j < 4; ++j)
                    a[ms].u[j] = AuU[rbase + ((kbase + 2 * j) >> 1)];
                #pragma unroll
                for (int j = 0; j < 4; ++j)
                    a[ms].u[4 + j] = AuU[rbase + ((kbase + 8 + 2 * j) >> 1)];
            }
            const int kcg = ch * 8 + kc;
            #pragma unroll
            for (int q = 0; q < 4; ++q) {
                FragBF b;
                const unsigned int* bp =
                    wsB0 + ((size_t)(((wave * 4 + q) * 64 + kcg) * 32 + lane)) * 8;
                #pragma unroll
                for (int j = 0; j < 8; ++j) b.u[j] = bp[j];
                #pragma unroll
                for (int ms = 0; ms < 4; ++ms)
                    acc[ms][q] = __builtin_amdgcn_wmma_f32_16x16x32_bf16(
                        false, a[ms].v, false, b.v, (short)0, acc[ms][q], false, false);
            }
        }
        __syncthreads();
    }

    // epilogue: per 16-row group, stage h (bf16) in LDS, run 512->32 layer
    unsigned int*   hU   = AuU;              // reuse A buffer (16*HW = 4160 dw)
    unsigned short* hS16 = (unsigned short*)AuU;
    for (int grp = 0; grp < 4; ++grp) {
        #pragma unroll
        for (int q = 0; q < 4; ++q) {
            int ng = (wave * 4 + q) * 16 + mrow;
            float bias = if0_b[ng];
            #pragma unroll
            for (int i = 0; i < 8; ++i) {
                int m = i + 8 * sel;         // row within group
                hS16[m * (2 * HW) + ng] = f2bf(eluf(acc[grp][q][i] + bias));
            }
        }
        __syncthreads();

        if (wave < 2) {                      // waves 0,1: one 16-wide N tile each
            v8f c2 = (v8f){0, 0, 0, 0, 0, 0, 0, 0};
            for (int kc = 0; kc < 16; ++kc) {
                FragBF a;
                const int kbase = kc * 32 + sel * 16;
                #pragma unroll
                for (int j = 0; j < 4; ++j)
                    a.u[j] = hU[mrow * HW + ((kbase + 2 * j) >> 1)];
                #pragma unroll
                for (int j = 0; j < 4; ++j)
                    a.u[4 + j] = hU[mrow * HW + ((kbase + 8 + 2 * j) >> 1)];
                FragBF b;
                const unsigned int* bp =
                    wsB1 + ((size_t)((wave * 16 + kc) * 32 + lane)) * 8;
                #pragma unroll
                for (int j = 0; j < 8; ++j) b.u[j] = bp[j];
                c2 = __builtin_amdgcn_wmma_f32_16x16x32_bf16(
                    false, a.v, false, b.v, (short)0, c2, false, false);
            }
            int ng = wave * 16 + mrow;
            float bias = if1_b[ng];
            #pragma unroll
            for (int i = 0; i < 8; ++i) {
                int m  = i + 8 * sel;
                int rb = grp * 16 + m;       // row within block
                P[((size_t)blockIdx.x * 64 + rb) * 32 + ng] =
                    c2[i] + bias + dirf[rb * 32 + ng];
            }
        }
        __syncthreads();
    }
}

// ---------------------------------------------------------------------------
// Kernel S: per-point view fusion (all small dims, VALU).  64 threads/point,
// 4 points per 256-thread block.
// ---------------------------------------------------------------------------
__global__ __launch_bounds__(256) void stage_b(
    const float* __restrict__ tok, const unsigned char* __restrict__ invalid,
    const float* __restrict__ P,
    const float* __restrict__ nr0_w, const float* __restrict__ nr0_b,
    const float* __restrict__ nr1_w, const float* __restrict__ nr1_b,
    const float* __restrict__ bf0_w, const float* __restrict__ bf0_b,
    const float* __restrict__ bf1_w, const float* __restrict__ bf1_b,
    const float* __restrict__ vf0_w, const float* __restrict__ vf0_b,
    const float* __restrict__ vf1_w, const float* __restrict__ vf1_b,
    const float* __restrict__ v20_w, const float* __restrict__ v20_b,
    const float* __restrict__ v21_w, const float* __restrict__ v21_b,
    const float* __restrict__ gf0_w, const float* __restrict__ gf0_b,
    const float* __restrict__ gf1_w, const float* __restrict__ gf1_b,
    float* __restrict__ out)
{
    __shared__ float tokS[4][8][16];
    __shared__ float bS[4][8][32];
    __shared__ float mS[4][8];
    __shared__ float wS[4][8];
    __shared__ float w0S[4][8];
    __shared__ float gfS[4][128];
    __shared__ float hS[4][8][64];
    __shared__ float xS[4][8][32];
    __shared__ float rS[4][8][33];
    __shared__ float visS[4][8];
    __shared__ float v2S[4][8];
    __shared__ float m2S[4][65];
    __shared__ float g1S[4][32];

    const int sub = threadIdx.x >> 6;
    const int t   = threadIdx.x & 63;
    const int pt  = blockIdx.x * 4 + sub;
    const size_t base = (size_t)pt * 8;

    for (int i = t; i < 128; i += 64) { int v = i >> 4, c = i & 15; tokS[sub][v][c] = tok[(base + v) * 16 + c]; }
    for (int i = t; i < 256; i += 64) { int v = i >> 5, c = i & 31; bS[sub][v][c] = P[(base + v) * 32 + c]; }
    if (t < 8) mS[sub][t] = invalid[base + t] ? 0.f : 1.f;
    __syncthreads();

    if (t < 8) {                                  // weights + nr MLP (16->8->1)
        int v = t;
        float s = 0.f;
        for (int u = 0; u < 8; ++u) s += mS[sub][u];
        float w = mS[sub][v] / (s + 1e-8f);
        wS[sub][v] = w;
        float h8[8];
        #pragma unroll
        for (int j = 0; j < 8; ++j) {
            float a = nr0_b[j];
            for (int i = 0; i < 16; ++i) a += tokS[sub][v][i] * nr0_w[i * 8 + j];
            h8[j] = eluf(a);
        }
        float o = nr1_b[0];
        #pragma unroll
        for (int j = 0; j < 8; ++j) o += h8[j] * nr1_w[j];
        w0S[sub][v] = sigm(o) * w;
    }
    __syncthreads();

    if (t < 32) {                                 // fused mean/var (x2)
        int c = t;
        float m0 = 0.f, m1 = 0.f;
        for (int v = 0; v < 8; ++v) { m0 += w0S[sub][v] * bS[sub][v][c]; m1 += wS[sub][v] * bS[sub][v][c]; }
        float v0 = 0.f, v1 = 0.f;
        for (int v = 0; v < 8; ++v) {
            float d0 = bS[sub][v][c] - m0; v0 += w0S[sub][v] * d0 * d0;
            float d1 = bS[sub][v][c] - m1; v1 += wS[sub][v] * d1 * d1;
        }
        gfS[sub][c] = m0; gfS[sub][32 + c] = v0; gfS[sub][64 + c] = m1; gfS[sub][96 + c] = v1;
    }
    __syncthreads();

    {                                             // bf0: 176 -> 64 (elu)
        int c = t;
        for (int v = 0; v < 8; ++v) {
            float a = bf0_b[c];
            for (int i = 0; i < 128; ++i) a += gfS[sub][i] * bf0_w[i * 64 + c];
            for (int i = 0; i < 32; ++i)  a += bS[sub][v][i] * bf0_w[(128 + i) * 64 + c];
            for (int i = 0; i < 16; ++i)  a += tokS[sub][v][i] * bf0_w[(160 + i) * 64 + c];
            hS[sub][v][c] = eluf(a);
        }
    }
    __syncthreads();

    if (t < 32) {                                 // bf1: 64 -> 32 (elu)
        int c = t;
        for (int v = 0; v < 8; ++v) {
            float a = bf1_b[c];
            for (int i = 0; i < 64; ++i) a += hS[sub][v][i] * bf1_w[i * 32 + c];
            xS[sub][v][c] = eluf(a);
        }
    }
    __syncthreads();

    if (t < 32) {                                 // vf0 on x*weight (elu) -> temp hS
        int c = t;
        for (int v = 0; v < 8; ++v) {
            float a = vf0_b[c], w = wS[sub][v];
            for (int i = 0; i < 32; ++i) a += xS[sub][v][i] * w * vf0_w[i * 32 + c];
            hS[sub][v][c] = eluf(a);
        }
    }
    __syncthreads();

    if (t < 33) {                                 // vf1: 32 -> 33 (elu)
        int c = t;
        for (int v = 0; v < 8; ++v) {
            float a = vf1_b[c];
            for (int i = 0; i < 32; ++i) a += hS[sub][v][i] * vf1_w[i * 33 + c];
            rS[sub][v][c] = eluf(a);
        }
    }
    __syncthreads();

    if (t < 32) { int c = t; for (int v = 0; v < 8; ++v) xS[sub][v][c] += rS[sub][v][c]; }
    if (t < 8)  { int v = t; visS[sub][v] = sigm(rS[sub][v][32]) * mS[sub][v]; }
    __syncthreads();

    if (t < 32) {                                 // v20 on x*vis (elu) -> temp hS
        int c = t;
        for (int v = 0; v < 8; ++v) {
            float a = v20_b[c], w = visS[sub][v];
            for (int i = 0; i < 32; ++i) a += xS[sub][v][i] * w * v20_w[i * 32 + c];
            hS[sub][v][c] = eluf(a);
        }
    }
    __syncthreads();

    if (t < 8) {                                  // v21: 32 -> 1, sigmoid * mask
        int v = t;
        float o = v21_b[0];
        for (int i = 0; i < 32; ++i) o += hS[sub][v][i] * v21_w[i];
        v2S[sub][v] = sigm(o) * mS[sub][v];
    }
    __syncthreads();

    if (t < 32) {                                 // final weighted mean/var over views
        int c = t;
        float s = 1e-8f;
        for (int v = 0; v < 8; ++v) s += v2S[sub][v];
        float wv[8], wsum = 0.f;
        for (int v = 0; v < 8; ++v) { wv[v] = v2S[sub][v] / s; wsum += wv[v]; }
        float m = 0.f;
        for (int v = 0; v < 8; ++v) m += wv[v] * xS[sub][v][c];
        float va = 0.f;
        for (int v = 0; v < 8; ++v) { float d = xS[sub][v][c] - m; va += wv[v] * d * d; }
        m2S[sub][c] = m; m2S[sub][32 + c] = va;
        if (c == 0) m2S[sub][64] = wsum * 0.125f;
    }
    __syncthreads();

    if (t < 32) {                                 // gf0: 65 -> 32 (elu)
        int c = t;
        float a = gf0_b[c];
        for (int i = 0; i < 65; ++i) a += m2S[sub][i] * gf0_w[i * 32 + c];
        g1S[sub][c] = eluf(a);
    }
    __syncthreads();

    if (t < 16) {                                 // gf1: 32 -> 16 (elu) -> out
        int c = t;
        float a = gf1_b[c];
        for (int i = 0; i < 32; ++i) a += g1S[sub][i] * gf1_w[i * 16 + c];
        out[(size_t)pt * 16 + c] = eluf(a);
    }
}

extern "C" void kernel_launch(void* const* d_in, const int* in_sizes, int n_in,
                              void* d_out, int out_size, void* d_ws, size_t ws_size,
                              hipStream_t stream)
{
    const float* tok  = (const float*)d_in[0];
    const float* bott = (const float*)d_in[1];
    const float* rayd = (const float*)d_in[2];
    const unsigned char* inval = (const unsigned char*)d_in[3];
    const float* rd0_w = (const float*)d_in[4],  *rd0_b = (const float*)d_in[5];
    const float* rd1_w = (const float*)d_in[6],  *rd1_b = (const float*)d_in[7];
    const float* if0_w = (const float*)d_in[8],  *if0_b = (const float*)d_in[9];
    const float* if1_w = (const float*)d_in[10], *if1_b = (const float*)d_in[11];
    const float* nr0_w = (const float*)d_in[12], *nr0_b = (const float*)d_in[13];
    const float* nr1_w = (const float*)d_in[14], *nr1_b = (const float*)d_in[15];
    const float* bf0_w = (const float*)d_in[16], *bf0_b = (const float*)d_in[17];
    const float* bf1_w = (const float*)d_in[18], *bf1_b = (const float*)d_in[19];
    const float* vf0_w = (const float*)d_in[20], *vf0_b = (const float*)d_in[21];
    const float* vf1_w = (const float*)d_in[22], *vf1_b = (const float*)d_in[23];
    const float* v20_w = (const float*)d_in[24], *v20_b = (const float*)d_in[25];
    const float* v21_w = (const float*)d_in[26], *v21_b = (const float*)d_in[27];
    const float* gf0_w = (const float*)d_in[28], *gf0_b = (const float*)d_in[29];
    const float* gf1_w = (const float*)d_in[30], *gf1_b = (const float*)d_in[31];

    unsigned int* wsB0 = (unsigned int*)d_ws;            // 2048*512 bf16 = 2 MB
    unsigned int* wsB1 = wsB0 + (2048 * 512 / 2);        // 512*32 bf16 = 32 KB
    float* P = (float*)(wsB1 + (512 * 32 / 2));          // 131072*32 f32 = 16 MB

    prep_weights<<<260, 256, 0, stream>>>(if0_w, if1_w, wsB0, wsB1);
    big_gemm<<<2048, 256, 0, stream>>>(bott, rayd, rd0_w, rd0_b, rd1_w, rd1_b,
                                       if0_b, if1_b, wsB0, wsB1, P);
    stage_b<<<4096, 256, 0, stream>>>(tok, inval, P,
                                      nr0_w, nr0_b, nr1_w, nr1_b,
                                      bf0_w, bf0_b, bf1_w, bf1_b,
                                      vf0_w, vf0_b, vf1_w, vf1_b,
                                      v20_w, v20_b, v21_w, v21_b,
                                      gf0_w, gf0_b, gf1_w, gf1_b,
                                      (float*)d_out);
}
